// Classifier_27092653703484
// MI455X (gfx1250) — matrix operand order repaired and checked by reading, not compiled, not run on hardware
//
#include <hip/hip_runtime.h>
#include <hip/hip_bf16.h>

// ---- CDNA5 WMMA types ----
typedef __attribute__((ext_vector_type(16))) __bf16 v16bf;
typedef __attribute__((ext_vector_type(8)))  float  v8f;

#define HIDDEN   128
#define NTILES   8           // 128 / 16 output-column tiles
#define FRAGS_PH 32          // 4 K-blocks x 8 N-tiles per phase
// LDS: [sel(hi,lo)][frag 0..31][lane 0..31][16 bf16]  = 2*32*32*16*2B = 64 KB
#define LDS_HALVES (2 * FRAGS_PH * 32 * 16)

static __device__ __forceinline__ __bf16 f2bf(float x) { return (__bf16)x; }

__global__ __launch_bounds__(256)
void edge_mlp_wmma(const float* __restrict__ uf,
                   const float* __restrict__ mf,
                   const int*   __restrict__ eidx,
                   const float* __restrict__ W1,
                   const float* __restrict__ b1,
                   const float* __restrict__ W2,
                   const float* __restrict__ b2,
                   float* __restrict__ out,
                   int E)
{
    __shared__ __bf16 sW[LDS_HALVES];

    const int tid  = threadIdx.x;
    const int wave = tid >> 5;
    const int lane = tid & 31;
    const int n15  = lane & 15;   // N column within 16-wide tile / A row
    const int kh   = lane >> 4;   // K-half selector (this code's contiguous k = 16*kh + j order)

    // Each wave owns one 32-edge tile (two M=16 sub-tiles).
    const long tileIdx = (long)blockIdx.x * 8 + wave;
    const long base_e  = tileIdx * 32;

    // Per-lane gather bases for the two M=16 sub-tiles (lanes 16-31 mirror 0-15).
    const float* up[2];
    const float* mp[2];
    #pragma unroll
    for (int t = 0; t < 2; ++t) {
        long e = base_e + t * 16 + n15;
        if (e >= E) e = (long)E - 1;            // clamp; stores are guarded below
        const int s = eidx[e];
        const int d = eidx[(long)E + e];
        up[t] = uf + (long)s * HIDDEN;
        mp[t] = mf + (long)d * HIDDEN;
    }

    // f32 accumulators: [m-tile][n-tile], 16x16 each -> 128 VGPRs
    v8f c[2][NTILES];
    #pragma unroll
    for (int t = 0; t < 2; ++t)
        #pragma unroll
        for (int n = 0; n < NTILES; ++n)
            c[t][n] = (v8f){0.f,0.f,0.f,0.f,0.f,0.f,0.f,0.f};

    const v16bf* sB = (const v16bf*)sW;

    // Two K-phases: phase 0 = K 0..127 (user half of W1), phase 1 = K 128..255 (movie half).
    #pragma unroll
    for (int p = 0; p < 2; ++p) {
        if (p) __syncthreads();   // LDS reuse: previous phase compute done

        // --- Cooperative repack: W1 -> split-bf16 (hi,lo) fragments in B lane layout ---
        // frag fl = kb_local*8 + nt ; element j of lane: K = kb*32 + (lane>>4)*16 + j,
        // N = nt*16 + (lane&15). Same contiguous k order as the A fragments.
        for (int idx = tid; idx < FRAGS_PH * 512; idx += 256) {
            const int j  = idx & 15;
            const int lr = (idx >> 4) & 31;
            const int fl = idx >> 9;
            const int kb = (p << 2) + (fl >> 3);
            const int nt = fl & 7;
            const int k  = (kb << 5) + ((lr >> 4) << 4) + j;
            const int n  = (nt << 4) + (lr & 15);
            const float w  = W1[k * HIDDEN + n];
            const __bf16 h = f2bf(w);
            sW[idx]                   = h;
            sW[FRAGS_PH * 512 + idx]  = f2bf(w - (float)h);
        }
        __syncthreads();

        // --- GEMM over this phase's 128 K values ---
        #pragma unroll
        for (int kbl = 0; kbl < 4; ++kbl) {
            // A fragments (16-bit A 16x32 layout): lane holds K run
            // [16*kh .. 16*kh+15] within the K-block.
            v16bf ah[2], al[2];
            #pragma unroll
            for (int t = 0; t < 2; ++t) {
                const float* bp = (p == 0) ? (up[t] + (kbl << 5))
                                           : (mp[t] + (kbl << 5));
                const float4 q0 = *(const float4*)(bp + 16 * kh);
                const float4 q1 = *(const float4*)(bp + 16 * kh + 4);
                const float4 q2 = *(const float4*)(bp + 16 * kh + 8);
                const float4 q3 = *(const float4*)(bp + 16 * kh + 12);
                const float fv[16] = {q0.x,q0.y,q0.z,q0.w, q1.x,q1.y,q1.z,q1.w,
                                      q2.x,q2.y,q2.z,q2.w, q3.x,q3.y,q3.z,q3.w};
                v16bf hi, lo;
                #pragma unroll
                for (int j = 0; j < 16; ++j) {
                    const float  x = fv[j];
                    const __bf16 h = f2bf(x);
                    hi[j] = h;
                    lo[j] = f2bf(x - (float)h);
                }
                ah[t] = hi; al[t] = lo;
            }

            #pragma unroll
            for (int nt = 0; nt < NTILES; ++nt) {
                const int fl = (kbl << 3) + nt;
                const v16bf bh = sB[(fl << 5) + lane];
                const v16bf bl = sB[FRAGS_PH * 32 + (fl << 5) + lane];
                #pragma unroll
                for (int t = 0; t < 2; ++t) {
                    // split-bf16: hi*hi + hi*lo + lo*hi  (~f32 precision)
                    c[t][nt] = __builtin_amdgcn_wmma_f32_16x16x32_bf16(
                        false, ah[t], false, bh, (short)0, c[t][nt], false, false);
                    c[t][nt] = __builtin_amdgcn_wmma_f32_16x16x32_bf16(
                        false, ah[t], false, bl, (short)0, c[t][nt], false, false);
                    c[t][nt] = __builtin_amdgcn_wmma_f32_16x16x32_bf16(
                        false, al[t], false, bh, (short)0, c[t][nt], false, false);
                }
            }
        }
    }

    // --- Epilogue: out = relu(C + b1) @ W2 + b2, all in f32 ---
    float b1v[NTILES], w2v[NTILES];
    #pragma unroll
    for (int nt = 0; nt < NTILES; ++nt) {
        const int n = (nt << 4) + n15;
        b1v[nt] = b1[n];
        w2v[nt] = W2[n];
    }
    const float bias2 = b2[0];

    // C layout: VGPR r, lanes 0-15 -> row r, lanes 16-31 -> row r+8; N = nt*16 + n15.
    float acc[2][8];
    #pragma unroll
    for (int t = 0; t < 2; ++t)
        #pragma unroll
        for (int r = 0; r < 8; ++r) {
            float a = 0.f;
            #pragma unroll
            for (int nt = 0; nt < NTILES; ++nt)
                a += fmaxf(c[t][nt][r] + b1v[nt], 0.f) * w2v[nt];
            acc[t][r] = a;
        }

    // Sum over the 16 N-columns held within each lane half.
    #pragma unroll
    for (int m = 1; m < 16; m <<= 1)
        #pragma unroll
        for (int t = 0; t < 2; ++t)
            #pragma unroll
            for (int r = 0; r < 8; ++r)
                acc[t][r] += __shfl_xor(acc[t][r], m, 32);

    // Lane 0 stores rows 0..7, lane 16 stores rows 8..15, per sub-tile.
    if (n15 == 0) {
        #pragma unroll
        for (int t = 0; t < 2; ++t) {
            const long e0 = base_e + t * 16 + kh * 8;
            if (e0 + 7 < E) {
                float4 o0 = {acc[t][0] + bias2, acc[t][1] + bias2,
                             acc[t][2] + bias2, acc[t][3] + bias2};
                float4 o1 = {acc[t][4] + bias2, acc[t][5] + bias2,
                             acc[t][6] + bias2, acc[t][7] + bias2};
                *(float4*)(out + e0)     = o0;
                *(float4*)(out + e0 + 4) = o1;
            } else {
                #pragma unroll
                for (int r = 0; r < 8; ++r)
                    if (e0 + r < E) out[e0 + r] = acc[t][r] + bias2;
            }
        }
    }
}

extern "C" void kernel_launch(void* const* d_in, const int* in_sizes, int n_in,
                              void* d_out, int out_size, void* d_ws, size_t ws_size,
                              hipStream_t stream) {
    const float* uf = (const float*)d_in[0];
    const float* mf = (const float*)d_in[1];
    const int*   ei = (const int*)  d_in[2];
    const float* W1 = (const float*)d_in[3];
    const float* b1 = (const float*)d_in[4];
    const float* W2 = (const float*)d_in[5];
    const float* b2 = (const float*)d_in[6];
    float* out = (float*)d_out;

    const int E      = in_sizes[2] / 2;        // edge_index is [2, E]
    const int tiles  = (E + 31) / 32;          // 32 edges per wave
    const int blocks = (tiles + 7) / 8;        // 8 waves (256 threads) per block

    edge_mlp_wmma<<<blocks, 256, 0, stream>>>(uf, mf, ei, W1, b1, W2, b2, out, E);
}
